// CTRGC_57458072485979
// MI455X (gfx1250) — matrix operand order repaired and checked by reading, not compiled, not run on hardware
//
#include <hip/hip_runtime.h>

typedef __bf16 bf16_t;
typedef __attribute__((ext_vector_type(16))) bf16_t v16bf;
typedef __attribute__((ext_vector_type(8)))  float  v8f;

#define NB   4
#define NN   32
#define CIN  64
#define COUT 64
#define CREL 8
#define TT   256
#define VV   25

// A-fragment (16-bit, 16x32 MxK) K index for element j in lane's v16bf:
// lanes 0-15: j=0..7 -> K=j,      j=8..15 -> K=16+(j-8)
// lanes 16-31: j=0..7 -> K=8+j,   j=8..15 -> K=24+(j-8)
__device__ __forceinline__ int ka_idx(int j, int lane) {
  return ((j & 8) << 1) + ((lane & 16) >> 1) + (j & 7);
}

// ---------------------------------------------------------------------------
// Kernel A: xm[b][n][c][v] = mean over T of x_b[n][c][t][v]
// ---------------------------------------------------------------------------
__global__ __launch_bounds__(256) void k_mean(
    const float* __restrict__ x0, const float* __restrict__ x1,
    const float* __restrict__ x2, const float* __restrict__ x3,
    float* __restrict__ xm) {
  int idx = blockIdx.x * blockDim.x + threadIdx.x;
  const int total = NB * NN * CIN * VV;
  if (idx >= total) return;
  int b  = idx / (NN * CIN * VV);
  int r  = idx - b * (NN * CIN * VV);
  int n  = r / (CIN * VV);
  int r2 = r - n * (CIN * VV);
  int c  = r2 / VV;
  int v  = r2 - c * VV;
  const float* x = (b == 0) ? x0 : (b == 1) ? x1 : (b == 2) ? x2 : x3;
  size_t base = ((size_t)(n * CIN + c) * TT) * VV + v;
  float s = 0.0f;
#pragma unroll 8
  for (int t = 0; t < TT; ++t) s += x[base + (size_t)t * VV];
  xm[idx] = s * (1.0f / (float)TT);
}

// ---------------------------------------------------------------------------
// Kernel B: build a_pad[b][n][c][u(32)][v(32)] (bf16, zero padded)
//   a = W4 . tanh(x1 - x2^T) + b4 + A    (alpha == 1.0)
// Layout chosen so stage-2 B-fragments are two contiguous b128 loads per lane.
// ---------------------------------------------------------------------------
__global__ __launch_bounds__(256) void k_att(
    const float* __restrict__ xm, const float* __restrict__ Ag,
    const float* __restrict__ W1, const float* __restrict__ B1,
    const float* __restrict__ W2, const float* __restrict__ B2,
    const float* __restrict__ W4, const float* __restrict__ B4,
    bf16_t* __restrict__ a_pad) {
  __shared__ float xs[CIN * VV];          // 1600 f32
  __shared__ float x1s[CREL * VV];        // 200
  __shared__ float x2s[CREL * VV];        // 200
  __shared__ float atts[CREL * VV * VV];  // 5000

  int tid = threadIdx.x;
  int b = blockIdx.x / NN;
  int n = blockIdx.x - b * NN;

  const float* xmb = xm + (size_t)(b * NN + n) * CIN * VV;
  for (int i = tid; i < CIN * VV; i += 256) xs[i] = xmb[i];
  __syncthreads();

  const float* w1 = W1 + b * CREL * CIN;  const float* b1 = B1 + b * CREL;
  const float* w2 = W2 + b * CREL * CIN;  const float* b2 = B2 + b * CREL;
  if (tid < CREL * VV) {
    int rr = tid / VV, u = tid - rr * VV;
    float s1 = b1[rr], s2 = b2[rr];
    for (int c = 0; c < CIN; ++c) {
      float xv = xs[c * VV + u];
      s1 += w1[rr * CIN + c] * xv;
      s2 += w2[rr * CIN + c] * xv;
    }
    x1s[tid] = s1;
    x2s[tid] = s2;
  }
  __syncthreads();

  for (int i = tid; i < CREL * VV * VV; i += 256) {
    int rr = i / (VV * VV);
    int uv = i - rr * VV * VV;
    int u = uv / VV, v = uv - (uv / VV) * VV;
    atts[i] = tanhf(x1s[rr * VV + u] - x2s[rr * VV + v]);
  }
  __syncthreads();

  const float* w4  = W4 + b * COUT * CREL;
  const float* b4  = B4 + b * COUT;
  const float* adj = Ag + b * VV * VV;
  bf16_t* ap = a_pad + ((size_t)(b * NN + n) * COUT << 10);
  for (int i = tid; i < COUT * 1024; i += 256) {
    int o = i >> 10;
    int u = (i >> 5) & 31;
    int v = i & 31;
    float acc = 0.0f;
    if (u < VV && v < VV) {
      acc = b4[o] + adj[u * VV + v];
      int uv = u * VV + v;
#pragma unroll
      for (int rr = 0; rr < CREL; ++rr)
        acc += w4[o * CREL + rr] * atts[rr * VV * VV + uv];
    }
    ap[i] = (bf16_t)acc;
  }
}

// ---------------------------------------------------------------------------
// Kernel C (fused main): per WG = (branch, n, 16-row t-block)
//  Stage 1: x3[o,t,v] = W3.x + b3  via WMMA (W3 A-frags in regs, x streamed
//           coalesced from global into B-frags), result -> LDS (bf16, v-padded)
//  Stage 2: out[t,u] = sum_v x3[t,v] * a[u,v]  via WMMA (A from LDS, B = a_pad)
// ---------------------------------------------------------------------------
__global__ __launch_bounds__(256) void k_main(
    const float* __restrict__ x0, const float* __restrict__ x1,
    const float* __restrict__ x2, const float* __restrict__ x3,
    const float* __restrict__ W3, const float* __restrict__ B3,
    const bf16_t* __restrict__ a_pad, float* __restrict__ out) {
  __shared__ bf16_t sX3[COUT][16][32];  // 65536 B, [c][t_local][v padded to 32]
  __shared__ float  sB3[COUT];

  int tid  = threadIdx.x;
  int lane = tid & 31;
  int wv   = tid >> 5;

  int blk    = blockIdx.x;
  int branch = blk >> 9;       // / (NN*16)
  int rem    = blk & 511;
  int n      = rem >> 4;
  int tblk   = rem & 15;
  int t0     = tblk * 16;

  const float* xg = (branch == 0) ? x0 : (branch == 1) ? x1 : (branch == 2) ? x2 : x3;
  const float* xb = xg + (size_t)n * CIN * TT * VV;
  const float* w3 = W3 + branch * COUT * CIN;

  // zero sX3 (only v=25..31 padding must be 0, but full clear is simplest)
  {
    unsigned int* p = (unsigned int*)&sX3[0][0][0];
    for (int i = tid; i < (COUT * 16 * 32) / 2; i += 256) p[i] = 0u;
    if (tid < COUT) sB3[tid] = B3[branch * COUT + tid];
  }

  // W3 A-fragments (4 M-tiles x 2 K-tiles), reused across all 25 N-tiles
  v16bf aW3[4][2];
#pragma unroll
  for (int mt = 0; mt < 4; ++mt) {
#pragma unroll
    for (int kk = 0; kk < 2; ++kk) {
      int o = mt * 16 + (lane & 15);
#pragma unroll
      for (int j = 0; j < 16; ++j)
        aW3[mt][kk][j] = (bf16_t)w3[o * CIN + kk * 32 + ka_idx(j, lane)];
    }
  }
  __syncthreads();

  // ---- Stage 1: x3 = W3 . x  (25 N-tiles of 16 (t,v)-columns each) ----
  for (int nt = wv; nt < 25; nt += 8) {
    int col = nt * 16 + (lane & 15);
    int tl  = col / 25;
    int vv  = col - tl * 25;
    const float* xcol = xb + (size_t)(t0 + tl) * VV + vv;

    if (nt + 8 < 25) {  // prefetch next tile's column (global_prefetch_b8)
      int coln = (nt + 8) * 16 + (lane & 15);
      int tln  = coln / 25;
      int vvn  = coln - tln * 25;
      __builtin_prefetch(xb + (size_t)(t0 + tln) * VV + vvn, 0, 1);
    }

    v16bf bx[2];
#pragma unroll
    for (int kk = 0; kk < 2; ++kk) {
      int c0 = kk * 32;  // B-frag K-tile base (K order per ka_idx)
#pragma unroll
      for (int j = 0; j < 16; ++j)
        bx[kk][j] = (bf16_t)xcol[(size_t)(c0 + ka_idx(j, lane)) * (TT * VV)];
    }

#pragma unroll
    for (int mt = 0; mt < 4; ++mt) {
      v8f acc = {0.f, 0.f, 0.f, 0.f, 0.f, 0.f, 0.f, 0.f};
      acc = __builtin_amdgcn_wmma_f32_16x16x32_bf16(false, aW3[mt][0], false, bx[0],
                                                    (short)0, acc, false, false);
      acc = __builtin_amdgcn_wmma_f32_16x16x32_bf16(false, aW3[mt][1], false, bx[1],
                                                    (short)0, acc, false, false);
      int obase = mt * 16 + ((lane & 16) >> 1);
#pragma unroll
      for (int i = 0; i < 8; ++i) {
        int o = obase + i;
        sX3[o][tl][vv] = (bf16_t)(acc[i] + sB3[o]);
      }
    }
  }
  __syncthreads();

  // ---- Stage 2: out[t,u] = sum_v x3[t,v] * a[u,v] ----
  size_t nc_base = (size_t)(branch * NN + n) * COUT;
  for (int ci = 0; ci < 8; ++ci) {
    int c = wv + ci * 8;

    // A-fragment: rows of sX3[c] (K = v, padded to 32)
    int trow = lane & 15;
    int off0 = (lane & 16);
    const bf16_t* pa = &sX3[c][trow][0];
    v16bf af;
#pragma unroll
    for (int j = 0; j < 8; ++j) af[j] = pa[off0 + j];
#pragma unroll
    for (int j = 0; j < 8; ++j) af[8 + j] = pa[8 + off0 + j];

    const bf16_t* ac = a_pad + ((nc_base + c) << 10);
#pragma unroll
    for (int un = 0; un < 2; ++un) {
      // B-fragment: lane = column u, contiguous 16-element v-slice
      const bf16_t* bp = ac + (size_t)(un * 16 + (lane & 15)) * 32 + (lane & 16);
      v16bf bfr;
#pragma unroll
      for (int j = 0; j < 16; ++j) bfr[j] = bp[j];

      v8f acc = {0.f, 0.f, 0.f, 0.f, 0.f, 0.f, 0.f, 0.f};
      acc = __builtin_amdgcn_wmma_f32_16x16x32_bf16(false, af, false, bfr,
                                                    (short)0, acc, false, false);

      int u = un * 16 + (lane & 15);
      if (u < VV) {
        float* op = out + ((nc_base + c) * TT + t0) * VV + u;
        int tb = (lane & 16) >> 1;
#pragma unroll
        for (int i = 0; i < 8; ++i) op[(size_t)(i + tb) * VV] = acc[i];
      }
    }
  }
}

// ---------------------------------------------------------------------------
extern "C" void kernel_launch(void* const* d_in, const int* in_sizes, int n_in,
                              void* d_out, int out_size, void* d_ws, size_t ws_size,
                              hipStream_t stream) {
  const float* jo = (const float*)d_in[0];
  const float* bo = (const float*)d_in[1];
  const float* jm = (const float*)d_in[2];
  const float* bm = (const float*)d_in[3];
  const float* A  = (const float*)d_in[4];
  const float* W1 = (const float*)d_in[5];
  const float* B1 = (const float*)d_in[6];
  const float* W2 = (const float*)d_in[7];
  const float* B2 = (const float*)d_in[8];
  const float* W3 = (const float*)d_in[9];
  const float* B3 = (const float*)d_in[10];
  const float* W4 = (const float*)d_in[11];
  const float* B4 = (const float*)d_in[12];
  float* out = (float*)d_out;

  // workspace: xm (819200 B, 256B aligned) then a_pad (bf16, 16 MB)
  float*  xm    = (float*)d_ws;
  bf16_t* a_pad = (bf16_t*)((char*)d_ws + (size_t)NB * NN * CIN * VV * sizeof(float));

  const int meanN = NB * NN * CIN * VV;
  k_mean<<<(meanN + 255) / 256, 256, 0, stream>>>(jo, bo, jm, bm, xm);
  k_att<<<NB * NN, 256, 0, stream>>>(xm, A, W1, B1, W2, B2, W4, B4, a_pad);
  k_main<<<NB * NN * 16, 256, 0, stream>>>(jo, bo, jm, bm, W3, B3, a_pad, out);
}
